// SynapseModel_70677981823275
// MI455X (gfx1250) — matrix operand order repaired and checked by reading, not compiled, not run on hardware
//
#include <hip/hip_runtime.h>
#include <hip/hip_bf16.h>
#include <math.h>

typedef _Float16 half_t;
typedef __attribute__((ext_vector_type(16))) _Float16 v16h;
typedef __attribute__((ext_vector_type(8)))  _Float16 v8h;
typedef __attribute__((ext_vector_type(8)))  float    v8f;

#define BATCH_M 8192
#define MT 256              // block M tile (8 waves x 32)
#define NT 64               // block N tile (4 sub-tiles of 16 per wave)
#define KC 128              // K chunk staged in LDS per double-buffer slot
#define LROW (KC + 8)       // padded LDS row stride (halves): 272B -> no bank conflict

__device__ __forceinline__ float gelu_exact(float x) {
    // torch nn.GELU default: 0.5*x*(1+erf(x/sqrt(2)))
    return 0.5f * x * (1.0f + erff(x * 0.70710678118654752440f));
}

__device__ __forceinline__ v16h ld_frag(const half_t* p) {
    v8h lo = *(const v8h*)(p);
    v8h hi = *(const v8h*)(p + 8);
    return __builtin_shufflevector(lo, hi, 0, 1, 2, 3, 4, 5, 6, 7,
                                   8, 9, 10, 11, 12, 13, 14, 15);
}

__device__ __forceinline__ v16h ld_afrag(const half_t* p) {
    // A fragment halves: [k0 .. k0+7] and [k0+16 .. k0+23] (per lane-half)
    v8h lo = *(const v8h*)(p);
    v8h hi = *(const v8h*)(p + 16);
    return __builtin_shufflevector(lo, hi, 0, 1, 2, 3, 4, 5, 6, 7,
                                   8, 9, 10, 11, 12, 13, 14, 15);
}

// ---------------------------------------------------------------------------
// Kernel 1: concat z|attention_out and convert f32 -> f16
// ---------------------------------------------------------------------------
__global__ void concat_f16_kernel(const float* __restrict__ z,
                                  const float* __restrict__ att,
                                  half_t* __restrict__ x,
                                  int total /* BATCH_M*2048 */) {
    for (int i = blockIdx.x * blockDim.x + threadIdx.x; i < total;
         i += gridDim.x * blockDim.x) {
        int c = i & 2047;
        int m = i >> 11;
        float v = (c < 1024) ? z[(m << 10) + c] : att[(m << 10) + (c - 1024)];
        x[i] = (half_t)v;
    }
}

// ---------------------------------------------------------------------------
// Kernel 2a: abs-sum reduction for absmean gamma (wave32 shuffle + atomicAdd)
// ---------------------------------------------------------------------------
__global__ void abssum_kernel(const float* __restrict__ W, int n,
                              float* __restrict__ acc) {
    float s = 0.0f;
    for (int i = blockIdx.x * blockDim.x + threadIdx.x; i < n;
         i += gridDim.x * blockDim.x)
        s += fabsf(W[i]);
    #pragma unroll
    for (int off = 16; off > 0; off >>= 1)
        s += __shfl_xor(s, off, 32);
    if ((threadIdx.x & 31) == 0)
        atomicAdd(acc, s);
}

// ---------------------------------------------------------------------------
// Kernel 2b: ternarize weights to f16: clip(round(W/gamma), -1, 1)
// ---------------------------------------------------------------------------
__global__ void ternarize_kernel(const float* __restrict__ W,
                                 const float* __restrict__ acc, float inv_n,
                                 half_t* __restrict__ out, int n) {
    float gamma = (*acc) * inv_n + 1e-8f;
    float inv_g = 1.0f / gamma;
    for (int i = blockIdx.x * blockDim.x + threadIdx.x; i < n;
         i += gridDim.x * blockDim.x) {
        float t = rintf(W[i] * inv_g);          // round-half-to-even == jnp.round
        t = fminf(1.0f, fmaxf(-1.0f, t));
        out[i] = (half_t)t;
    }
}

// ---------------------------------------------------------------------------
// Kernel 3: WMMA GEMM + bias + exact GELU
//   Y[M,N] = gelu(X[M,K] @ Wt[N,K]^T + b)
//   block = 256 threads (8 waves); block tile 256(M) x 64(N)
//   wave tile = 32(M) x 64(N): 8 accumulators, 8 WMMAs per (2xA, 4xB) loads
//   Wt chunk (64 rows x 128 K) staged via GLOBAL_LOAD_ASYNC_TO_LDS_B128,
//   double-buffered, synchronized with s_wait_asynccnt + barriers.
// ---------------------------------------------------------------------------
__global__ void __launch_bounds__(256)
bitlinear_gemm_kernel(const half_t* __restrict__ X,   // M x K (f16)
                      const half_t* __restrict__ Wt,  // N x K (f16 ternary)
                      const float* __restrict__ bias, // N
                      half_t* __restrict__ Yh,        // M x N f16 (intermediate)
                      float* __restrict__ Yf,         // M x N f32 (final) or null
                      int K, int N) {
    __shared__ half_t ldsW[2][NT * LROW];             // 2 x 17 KB, padded rows

    const int tid  = threadIdx.x;
    const int wave = tid >> 5;
    const int lane = tid & 31;
    const int hid  = lane >> 4;                       // 0 or 1 (lane half)
    const int lm   = lane & 15;

    const int n0 = blockIdx.y * NT;
    const int m0 = blockIdx.x * MT + wave * 32;       // wave owns rows [m0, m0+32)

    const unsigned long long wbase = (unsigned long long)(const void*)Wt;

    // Issue the async DMA of Wt[n0 .. n0+NT) columns [c*KC, (c+1)*KC) into
    // LDS buffer `buf`. 64 rows x 16 b128 vectors = 1024 transfers, 4/thread.
    auto issue_chunk = [&](int c, int buf) {
        const unsigned lds0 =
            (unsigned)(unsigned long long)(const void*)&ldsW[buf][0];
        #pragma unroll
        for (int i = 0; i < (NT * KC / 8) / 256; ++i) {
            const int v   = tid + i * 256;
            const int row = v >> 4;                    // KC/8 == 16 vectors/row
            const int col = v & 15;
            const unsigned goff =
                (unsigned)((((size_t)(n0 + row)) * K + (size_t)c * KC + col * 8)
                           * sizeof(half_t));
            const unsigned ldsa =
                lds0 + (unsigned)(row * LROW + col * 8) * (unsigned)sizeof(half_t);
            asm volatile("global_load_async_to_lds_b128 %0, %1, %2"
                         :: "v"(ldsa), "v"(goff), "s"(wbase)
                         : "memory");
        }
    };

    // A fragment sources: lane covers rows m0+lm and m0+16+lm.
    //   lanes 0-15:  k0+[0..7],  k0+16+[0..7]
    //   lanes 16-31: k0+[8..15], k0+24+[0..7]
    const half_t* xrow0 = X + (size_t)(m0 + lm) * K + hid * 8;
    const half_t* xrow1 = xrow0 + (size_t)16 * K;

    v8f acc[4][2];
    #pragma unroll
    for (int j = 0; j < 4; ++j) {
        acc[j][0] = v8f{};
        acc[j][1] = v8f{};
    }

    const int nchunks = K / KC;
    issue_chunk(0, 0);

    for (int c = 0; c < nchunks; ++c) {
        const int buf = c & 1;
        asm volatile("s_wait_asynccnt 0x0" ::: "memory");
        __syncthreads();
        if (c + 1 < nchunks) issue_chunk(c + 1, buf ^ 1);

        __builtin_prefetch(xrow0 + (size_t)c * KC + 2 * KC, 0, 0);
        __builtin_prefetch(xrow1 + (size_t)c * KC + 2 * KC, 0, 0);

        const half_t* lb = &ldsW[buf][0];
        #pragma unroll
        for (int step = 0; step < KC / 32; ++step) {
            const int kg = c * KC + step * 32;
            // 2 A fragments (global), 4 B fragments (LDS) — all loads first
            v16h a0 = ld_afrag(xrow0 + kg);
            v16h a1 = ld_afrag(xrow1 + kg);
            v16h b0, b1, b2, b3;
            {
                const half_t* bbase = lb + step * 32 + hid * 8 + (size_t)lm * LROW;
                b0 = ld_afrag(bbase);
                b1 = ld_afrag(bbase + 16 * LROW);
                b2 = ld_afrag(bbase + 32 * LROW);
                b3 = ld_afrag(bbase + 48 * LROW);
            }
            // 8 back-to-back WMMAs reusing both A and B fragments
            acc[0][0] = __builtin_amdgcn_wmma_f32_16x16x32_f16(
                false, a0, false, b0, (short)0, acc[0][0], false, false);
            acc[0][1] = __builtin_amdgcn_wmma_f32_16x16x32_f16(
                false, a1, false, b0, (short)0, acc[0][1], false, false);
            acc[1][0] = __builtin_amdgcn_wmma_f32_16x16x32_f16(
                false, a0, false, b1, (short)0, acc[1][0], false, false);
            acc[1][1] = __builtin_amdgcn_wmma_f32_16x16x32_f16(
                false, a1, false, b1, (short)0, acc[1][1], false, false);
            acc[2][0] = __builtin_amdgcn_wmma_f32_16x16x32_f16(
                false, a0, false, b2, (short)0, acc[2][0], false, false);
            acc[2][1] = __builtin_amdgcn_wmma_f32_16x16x32_f16(
                false, a1, false, b2, (short)0, acc[2][1], false, false);
            acc[3][0] = __builtin_amdgcn_wmma_f32_16x16x32_f16(
                false, a0, false, b3, (short)0, acc[3][0], false, false);
            acc[3][1] = __builtin_amdgcn_wmma_f32_16x16x32_f16(
                false, a1, false, b3, (short)0, acc[3][1], false, false);
        }
        __syncthreads();
    }

    // --- epilogue: bias + exact GELU ---------------------------------------
    // D layout: lane lm (+16*hid), VGPR r -> (m0 + 16*i + r + 8*hid, n0+16j+lm)
    const int mrow = m0 + (hid << 3);
    #pragma unroll
    for (int j = 0; j < 4; ++j) {
        const int n = n0 + 16 * j + lm;
        const float bv = bias[n];
        #pragma unroll
        for (int i = 0; i < 2; ++i) {
            #pragma unroll
            for (int r = 0; r < 8; ++r) {
                const int m = mrow + 16 * i + r;
                float g = gelu_exact(acc[j][i][r] + bv);
                if (Yf) {
                    Yf[(size_t)m * N + n] = g;
                } else {
                    Yh[(size_t)m * N + n] = (half_t)g;
                }
            }
        }
    }
}

// ---------------------------------------------------------------------------
// Host-side orchestration
// ---------------------------------------------------------------------------
static inline size_t align256(size_t x) { return (x + 255) & ~(size_t)255; }

extern "C" void kernel_launch(void* const* d_in, const int* in_sizes, int n_in,
                              void* d_out, int out_size, void* d_ws, size_t ws_size,
                              hipStream_t stream) {
    (void)in_sizes; (void)n_in; (void)out_size; (void)ws_size;

    const float* z   = (const float*)d_in[0];
    const float* att = (const float*)d_in[1];

    // Layer tables: We0, We1, We2, Wb, Wd0, Wd1, Wd2
    const float* W[7]  = {(const float*)d_in[2],  (const float*)d_in[4],
                          (const float*)d_in[6],  (const float*)d_in[8],
                          (const float*)d_in[10], (const float*)d_in[12],
                          (const float*)d_in[14]};
    const float* B[7]  = {(const float*)d_in[3],  (const float*)d_in[5],
                          (const float*)d_in[7],  (const float*)d_in[9],
                          (const float*)d_in[11], (const float*)d_in[13],
                          (const float*)d_in[15]};
    const int    Ks[7] = {2048, 1024, 512, 256, 256, 512, 1024};
    const int    Ns[7] = {1024, 512, 256, 256, 512, 1024, 2048};

    // ---- workspace layout --------------------------------------------------
    char* ws = (char*)d_ws;
    float* gsum = (float*)ws;                                   // 8 floats
    size_t off = 256;
    half_t* actA = (half_t*)(ws + off); off += align256((size_t)BATCH_M * 2048 * sizeof(half_t));
    half_t* actB = (half_t*)(ws + off); off += align256((size_t)BATCH_M * 2048 * sizeof(half_t));
    half_t* wt[7];
    for (int l = 0; l < 7; ++l) {
        size_t elems = (size_t)Ns[l] * Ks[l];
        wt[l] = (half_t*)(ws + off);
        off += align256(elems * sizeof(half_t));
    }

    // ---- 0) zero gamma accumulators ---------------------------------------
    hipMemsetAsync(gsum, 0, 8 * sizeof(float), stream);

    // ---- 1) concat + f16 convert ------------------------------------------
    {
        int total = BATCH_M * 2048;
        concat_f16_kernel<<<1024, 256, 0, stream>>>(z, att, actA, total);
    }

    // ---- 2) absmean + ternarize each weight matrix ------------------------
    for (int l = 0; l < 7; ++l) {
        int n = Ns[l] * Ks[l];
        int blocks = (n + 255) / 256; if (blocks > 512) blocks = 512;
        abssum_kernel<<<blocks, 256, 0, stream>>>(W[l], n, &gsum[l]);
    }
    for (int l = 0; l < 7; ++l) {
        int n = Ns[l] * Ks[l];
        int blocks = (n + 255) / 256; if (blocks > 512) blocks = 512;
        ternarize_kernel<<<blocks, 256, 0, stream>>>(W[l], &gsum[l],
                                                     1.0f / (float)n, wt[l], n);
    }

    // ---- 3) seven fused WMMA GEMM + bias + GELU layers --------------------
    half_t* xin = actA;
    half_t* xout = actB;
    for (int l = 0; l < 7; ++l) {
        const int K = Ks[l], N = Ns[l];
        dim3 grid(BATCH_M / MT, N / NT);
        const bool last = (l == 6);
        bitlinear_gemm_kernel<<<grid, 256, 0, stream>>>(
            xin, wt[l], B[l],
            last ? nullptr : xout,
            last ? (float*)d_out : nullptr,
            K, N);
        half_t* t = xin; xin = xout; xout = t;
    }
}
